// SSIM2_12730283065958
// MI455X (gfx1250) — matrix operand order repaired and checked by reading, not compiled, not run on hardware
//
#include <hip/hip_runtime.h>

typedef __attribute__((ext_vector_type(16))) _Float16 v16h;
typedef __attribute__((ext_vector_type(8)))  _Float16 v8h;
typedef __attribute__((ext_vector_type(4)))  _Float16 v4h;
typedef __attribute__((ext_vector_type(8)))  float    v8f;
typedef __attribute__((ext_vector_type(4)))  float    v4f;

#define IMG   512
#define TILE  16
#define LDW   40                 // H^T row stride in halves (80B, 16B-aligned)
#define HT_STRIDE (16 * LDW)     // halves per H^T buffer
#define WAVE_LDS  (5 * HT_STRIDE)
#define NWAVES 4
#define INV_N (1.0f / 25165824.0f)   // 1 / (32*3*512*512)

static __device__ __forceinline__ void lds_fence() {
    // wave-private LDS slices: DS ops are in-order per wave; this orders the
    // compiler and drains DScnt without an inter-wave s_barrier.
    asm volatile("s_wait_dscnt 0x0" ::: "memory");
}

// Load 8 consecutive f32 starting at (clamped) column sc of a clamped row,
// convert to f16, and scale by a 0/1 validity mask. Branch-free.
static __device__ __forceinline__ v8h load8(const float* rowp, int sc, _Float16 m) {
    int scc = min(max(sc, 0), IMG - 8);      // always a legal address
    v4f a = *(const v4f*)(rowp + scc);
    v4f b = *(const v4f*)(rowp + scc + 4);
    v8h h = __builtin_shufflevector(__builtin_convertvector(a, v4h),
                                    __builtin_convertvector(b, v4h),
                                    0, 1, 2, 3, 4, 5, 6, 7);
    return h * m;                            // v_pk_mul_f16, splat mask
}

__global__ void ssim_zero_out(float* out) {
    if (threadIdx.x < 3) out[threadIdx.x] = 0.0f;
}

__global__ __launch_bounds__(128)
void ssim_fused(const float* __restrict__ img1, const float* __restrict__ img2,
                float* __restrict__ out) {
    __shared__ __align__(16) _Float16 lds[NWAVES * WAVE_LDS];

    const int lane = threadIdx.x & 31;
    const int wave = threadIdx.x >> 5;
    _Float16* HTbase = &lds[wave * WAVE_LDS];   // 5 x (16 x LDW) H^T buffers

    const int tx = blockIdx.x * NWAVES + wave;  // tile col 0..31
    const int ty = blockIdx.y;                  // tile row 0..31
    const int plane = blockIdx.z;               // 0..95
    const int x0 = tx * TILE, y0 = ty * TILE;
    const float* p1 = img1 + (size_t)plane * IMG * IMG;
    const float* p2 = img2 + (size_t)plane * IMG * IMG;

    const int nl = lane & 15, half = lane >> 4;

    // ---- Banded Gaussian fragments via lane broadcast (no memory, no branches)
    // g[0..10] lives in lanes 0..10; lanes 11..31 hold 0.  Out-of-band indices
    // wrap (mod 32) onto lanes >= 11, yielding the required zero weights.
    float tq = (float)(lane - 5);
    float gl = (lane < 11) ? (__expf(-tq * tq * (1.0f / 4.5f)) * 0.26601171f)
                           : 0.0f;
    const int bb = 16 * half - nl - 3;   // band shift: window starts at x0-8
    const int ab = 8 * half - nl - 3;
    v16h Bh;  // horizontal B (32x16): Bh[k][n] = g[k-n-3]
    v16h Av;  // vertical   A (16x32): Av[m][k] = g[k-m-3]
    #pragma unroll
    for (int e = 0; e < 16; ++e) {
        Bh[e] = (_Float16)__shfl(gl, (bb + e) & 31, 32);
        int ia = (e < 8) ? (ab + e) : (ab + 8 + e);
        Av[e] = (_Float16)__shfl(gl, ia & 31, 32);
    }

    // ---- Horizontal pass: A-fragments loaded directly from global ----
    #pragma unroll
    for (int blk = 0; blk < 2; ++blk) {
        int r  = nl + 16 * blk;          // A layout: lane holds row M = lane&15
        int gr = y0 - 8 + r;
        int grc = min(max(gr, 0), IMG - 1);
        bool rowok = ((unsigned)gr < IMG);
        const float* r1 = p1 + (size_t)grc * IMG;
        const float* r2 = p2 + (size_t)grc * IMG;
        int s0 = x0 - 8 + 16 * half;     // K = 16*half .. +7   (8-aligned)
        int s1 = s0 + 8;                 // K = 16*half+8 .. +7
        _Float16 m0 = (_Float16)((rowok && (unsigned)s0 < IMG) ? 1.0f : 0.0f);
        _Float16 m1 = (_Float16)((rowok && (unsigned)s1 < IMG) ? 1.0f : 0.0f);
        v8h xa = load8(r1, s0, m0);
        v8h xb = load8(r1, s1, m1);
        v8h ya = load8(r2, s0, m0);
        v8h yb = load8(r2, s1, m1);
        #pragma unroll
        for (int f = 0; f < 5; ++f) {
            v8h lo, hi;
            switch (f) {
                case 0:  lo = xa;      hi = xb;      break;
                case 1:  lo = ya;      hi = yb;      break;
                case 2:  lo = xa * xa; hi = xb * xb; break;   // v_pk_mul_f16
                case 3:  lo = ya * ya; hi = yb * yb; break;
                default: lo = xa * ya; hi = xb * yb; break;
            }
            v16h A = __builtin_shufflevector(lo, hi, 0, 1, 2, 3, 4, 5, 6, 7,
                                             8, 9, 10, 11, 12, 13, 14, 15);
            v8f c = {};
            c = __builtin_amdgcn_wmma_f32_16x16x32_f16(false, A, false, Bh,
                                                       (short)0, c, false, false);
            // store transposed: C layout lane holds (M = v + 8*half + 16*blk, N = nl)
            v8h hh;
            #pragma unroll
            for (int v = 0; v < 8; ++v) hh[v] = (_Float16)c[v];
            _Float16* HT = HTbase + f * HT_STRIDE;
            *(v8h*)&HT[nl * LDW + 8 * half + 16 * blk] = hh;
        }
    }

    lds_fence();

    // ---- Vertical pass: one WMMA per field ----
    v8f R[5];
    #pragma unroll
    for (int f = 0; f < 5; ++f) {
        _Float16* HT = HTbase + f * HT_STRIDE;
        v8h b0 = *(const v8h*)&HT[nl * LDW + 8 * half];
        v8h b1 = *(const v8h*)&HT[nl * LDW + 8 * half + 16];
        v16h B = __builtin_shufflevector(b0, b1, 0, 1, 2, 3, 4, 5, 6, 7,
                                         8, 9, 10, 11, 12, 13, 14, 15);
        v8f c = {};
        R[f] = __builtin_amdgcn_wmma_f32_16x16x32_f16(false, Av, false, B,
                                                      (short)0, c, false, false);
    }

    // ---- Pointwise SSIM terms + mean reduction (fast rcp/sqrt) ----
    const float C1 = 1e-4f, C2 = 9e-4f, C3 = 4.5e-4f;
    float sL = 0.0f, sC = 0.0f, sS = 0.0f;
    #pragma unroll
    for (int v = 0; v < 8; ++v) {
        float m1 = R[0][v], m2 = R[1][v];
        float m1s = m1 * m1, m2s = m2 * m2, m12 = m1 * m2;
        float a1 = fabsf(R[2][v] - m1s);
        float a2 = fabsf(R[3][v] - m2s);
        float s12 = R[4][v] - m12;
        float q12 = __builtin_amdgcn_sqrtf(a1 * a2);   // sqrt(a1)*sqrt(a2)
        sL = fmaf(2.0f * m12 + C1, __builtin_amdgcn_rcpf(m1s + m2s + C1), sL);
        sC = fmaf(2.0f * q12 + C2, __builtin_amdgcn_rcpf(a1 + a2 + C2), sC);
        sS = fmaf(s12 + C3,        __builtin_amdgcn_rcpf(q12 + C3),     sS);
    }
    for (int off = 16; off > 0; off >>= 1) {
        sL += __shfl_xor(sL, off, 32);
        sC += __shfl_xor(sC, off, 32);
        sS += __shfl_xor(sS, off, 32);
    }
    if (lane == 0) {
        atomicAdd(&out[0], sL * INV_N);
        atomicAdd(&out[1], sC * INV_N);
        atomicAdd(&out[2], sS * INV_N);
    }
}

extern "C" void kernel_launch(void* const* d_in, const int* in_sizes, int n_in,
                              void* d_out, int out_size, void* d_ws, size_t ws_size,
                              hipStream_t stream) {
    const float* img1 = (const float*)d_in[0];
    const float* img2 = (const float*)d_in[1];
    float* out = (float*)d_out;

    ssim_zero_out<<<1, 64, 0, stream>>>(out);
    dim3 grid(32 / NWAVES, 32, 96);   // tile cols / waves-per-WG, tile rows, planes
    dim3 block(NWAVES * 32);
    ssim_fused<<<grid, block, 0, stream>>>(img1, img2, out);
}
